// UltimateLegionNet_11201274708161
// MI455X (gfx1250) — matrix operand order repaired and checked by reading, not compiled, not run on hardware
//
#include <hip/hip_runtime.h>
#include <hip/hip_bf16.h>

typedef __attribute__((ext_vector_type(16))) _Float16 v16h;
typedef __attribute__((ext_vector_type(8)))  _Float16 h8;
typedef __attribute__((ext_vector_type(8)))  float    v8f;

#define N_AG 8192
#define HDIM 128
#define SCALE_QK 0.08838834764831845f  /* 1/sqrt(128) */

// Flip to 0 if the assembler rejects the async-to-LDS mnemonics.
#define KV_ASYNC 1

// ---------------------------------------------------------------------------
// WMMA helper (gfx1250: 16x16x32 f16 -> f32)
// ---------------------------------------------------------------------------
__device__ __forceinline__ v8f wmma_f16(v16h a, v16h b, v8f c) {
  return __builtin_amdgcn_wmma_f32_16x16x32_f16(false, a, false, b, (short)0, c,
                                                false, false);
}

// Load a 16x32 f16 A-fragment for one lane from a row-major matrix.
// A-layout (ISA 7.12.2): lane l holds row M=l&15, K {base..base+7, base+16..
// base+23} with base=(l>=16)?8:0. p = A_row + base (16B aligned).
__device__ __forceinline__ v16h load_a16(const _Float16* p) {
  h8 c0 = *(const h8*)(p);
  h8 c1 = *(const h8*)(p + 16);
  v16h a;
#pragma unroll
  for (int i = 0; i < 8; ++i) { a[i] = c0[i]; a[i + 8] = c1[i]; }
  return a;
}

// ---------------------------------------------------------------------------
// Async global->LDS copy (CDNA5 TDM-lite path, ASYNCcnt-tracked), 16B per lane.
// Generic LDS pointer low 32 bits are the LDS byte offset (aperture layout).
// ---------------------------------------------------------------------------
__device__ __forceinline__ void copy_b128_to_lds(void* lds, const void* g) {
#if defined(__HIP_DEVICE_COMPILE__) && KV_ASYNC
  unsigned lo = (unsigned)(unsigned long long)lds;
  unsigned long long ga = (unsigned long long)g;
  asm volatile("global_load_async_to_lds_b128 %0, %1, off"
               :: "v"(lo), "v"(ga)
               : "memory");
#else
  *(int4*)lds = *(const int4*)g;
#endif
}

__device__ __forceinline__ void kv_copy_wait() {
#if defined(__HIP_DEVICE_COMPILE__) && KV_ASYNC
  asm volatile("s_wait_asynccnt 0x0" ::: "memory");
#endif
}

// ---------------------------------------------------------------------------
// Conv1: [N,3,40,40] -> relu -> [N,16,18,18] fp32
// ---------------------------------------------------------------------------
__global__ void conv1_relu(const float* __restrict__ img,
                           const float* __restrict__ w,
                           const float* __restrict__ b,
                           float* __restrict__ out) {
  int idx = blockIdx.x * blockDim.x + threadIdx.x;
  if (idx >= N_AG * 16 * 18 * 18) return;
  int ox = idx % 18; int t = idx / 18;
  int oy = t % 18;   t /= 18;
  int oc = t % 16;   int n = t / 16;
  float acc = b[oc];
  const float* ip = img + (size_t)n * 3 * 1600;
  const float* wp = w + oc * 75;
  int iy0 = oy * 2, ix0 = ox * 2;
#pragma unroll
  for (int ic = 0; ic < 3; ++ic)
#pragma unroll
    for (int ky = 0; ky < 5; ++ky)
#pragma unroll
      for (int kx = 0; kx < 5; ++kx)
        acc += ip[ic * 1600 + (iy0 + ky) * 40 + ix0 + kx] *
               wp[ic * 25 + ky * 5 + kx];
  out[idx] = fmaxf(acc, 0.f);
}

// ---------------------------------------------------------------------------
// Conv2: [N,16,18,18] -> relu -> flat [N,2048] f16 (matches NCHW reshape)
// ---------------------------------------------------------------------------
__global__ void conv2_relu_h(const float* __restrict__ x,
                             const float* __restrict__ w,
                             const float* __restrict__ b,
                             _Float16* __restrict__ out) {
  int idx = blockIdx.x * blockDim.x + threadIdx.x;
  if (idx >= N_AG * 2048) return;
  int ox = idx & 7;  int t = idx >> 3;
  int oy = t & 7;    t >>= 3;
  int oc = t & 31;   int n = t >> 5;
  float acc = b[oc];
  const float* ip = x + (size_t)n * 16 * 324;
  const float* wp = w + oc * 144;
  int iy0 = oy * 2, ix0 = ox * 2;
#pragma unroll
  for (int ic = 0; ic < 16; ++ic)
#pragma unroll
    for (int ky = 0; ky < 3; ++ky)
#pragma unroll
      for (int kx = 0; kx < 3; ++kx)
        acc += ip[ic * 324 + (iy0 + ky) * 18 + ix0 + kx] *
               wp[ic * 9 + ky * 3 + kx];
  out[idx] = (_Float16)fmaxf(acc, 0.f);
}

// ---------------------------------------------------------------------------
// Pack a [Nsrc,Ksrc] f32 weight into WMMA-B fragment order (f16, zero pad).
// Bp[((kt*Ntiles + tn)*32 + lane)*16 + s] = W[tn*16+(lane&15)][kt*32 + ks],
// ks = (lane>>4)*8 + (s&7) + (s>>3)*16.
// ---------------------------------------------------------------------------
__global__ void pack_wb(const float* __restrict__ W, _Float16* __restrict__ out,
                        int Nsrc, int Ksrc, int Ktiles, int Ntiles) {
  int idx = blockIdx.x * blockDim.x + threadIdx.x;
  if (idx >= Ktiles * Ntiles * 512) return;
  int s = idx & 15;
  int lane = (idx >> 4) & 31;
  int n = lane & 15;
  int tn = (idx >> 9) % Ntiles;
  int kt = (idx >> 9) / Ntiles;
  int k = kt * 32 + (lane >> 4) * 8 + (s & 7) + (s >> 3) * 16;
  int nc = tn * 16 + n;
  float v = (nc < Nsrc && k < Ksrc) ? W[(size_t)nc * Ksrc + k] : 0.f;
  out[idx] = (_Float16)v;
}

// Pack K [8192,128] f16 into score-B order: Kp[(jt*4+kc)*512 + lane*16 + n]
__global__ void pack_kb(const _Float16* __restrict__ Kh,
                        _Float16* __restrict__ Kp) {
  int idx = blockIdx.x * blockDim.x + threadIdx.x;
  if (idx >= 512 * 4 * 512) return;
  int s = idx & 15;
  int lane = (idx >> 4) & 31;
  int n = lane & 15;
  int kc = (idx >> 9) & 3;
  int jt = idx >> 11;
  int k = kc * 32 + (lane >> 4) * 8 + (s & 7) + (s >> 3) * 16;
  Kp[idx] = Kh[(size_t)(jt * 16 + n) * HDIM + k];
}

// Pack V [8192,128] f16 into context-B order: Vp[(jv*8+t)*512 + lane*16 + n]
__global__ void pack_vb(const _Float16* __restrict__ Vh,
                        _Float16* __restrict__ Vp) {
  int idx = blockIdx.x * blockDim.x + threadIdx.x;
  if (idx >= 256 * 8 * 512) return;
  int s = idx & 15;
  int lane = (idx >> 4) & 31;
  int n = lane & 15;
  int t = (idx >> 9) & 7;
  int jv = idx >> 12;
  int key = jv * 32 + (lane >> 4) * 8 + (s & 7) + (s >> 3) * 16;
  Vp[idx] = Vh[(size_t)key * HDIM + t * 16 + n];
}

// Pack adjacency int32 [8192,8192] -> bitmask u32 [8192,256] (coalesced b128)
__global__ void pack_adj(const int* __restrict__ adj,
                         unsigned int* __restrict__ amask) {
  int idx = blockIdx.x * blockDim.x + threadIdx.x;
  if (idx >= N_AG * 256) return;
  int m = idx >> 8, w = idx & 255;
  const int4* p = (const int4*)(adj + (size_t)m * N_AG + w * 32);
  unsigned int bits = 0;
#pragma unroll
  for (int g = 0; g < 8; ++g) {
    int4 v = p[g];
    bits |= (v.x != 0 ? 1u : 0u) << (g * 4 + 0);
    bits |= (v.y != 0 ? 1u : 0u) << (g * 4 + 1);
    bits |= (v.z != 0 ? 1u : 0u) << (g * 4 + 2);
    bits |= (v.w != 0 ? 1u : 0u) << (g * 4 + 3);
  }
  amask[idx] = bits;
}

// ---------------------------------------------------------------------------
// Misc elementwise
// ---------------------------------------------------------------------------
__global__ void f32_to_f16(const float* __restrict__ s, _Float16* __restrict__ d,
                           int count) {
  int idx = blockIdx.x * blockDim.x + threadIdx.x;
  if (idx < count) d[idx] = (_Float16)s[idx];
}

__global__ void embed_fill(const float* __restrict__ et,
                           const int* __restrict__ cls,
                           _Float16* __restrict__ feat) {
  int idx = blockIdx.x * blockDim.x + threadIdx.x;
  if (idx >= N_AG * 32) return;
  int e = idx & 31, n = idx >> 5;
  float v = 0.f;
  if (e < 16) v = et[cls[n] * 16 + e];
  feat[(size_t)n * 160 + 128 + e] = (_Float16)v;
}

// ---------------------------------------------------------------------------
// GEMM: out[M,N] = A[M,K] * W^T + bias, W pre-packed in B-fragment order.
// One 16x16 tile per wave, software-pipelined 16x16x32 f16 WMMA k-loop.
// Grid sized exactly: (Mtiles*Ntiles) waves, blockDim=128 (4 waves).
// ---------------------------------------------------------------------------
__global__ void gemm_xwt(const _Float16* __restrict__ A, int lda,
                         const _Float16* __restrict__ Bp,
                         const float* __restrict__ bias,
                         float* __restrict__ outF, int ldoF,
                         _Float16* __restrict__ outH, int ldoH,
                         int Ntiles, int K, int nval, int relu) {
  int wave = (blockIdx.x * blockDim.x + threadIdx.x) >> 5;
  int lane = threadIdx.x & 31;
  int tm = wave / Ntiles, tn = wave % Ntiles;
  int row = lane & 15, hi = lane >> 4;
  const _Float16* ap = A + (size_t)(tm * 16 + row) * lda + hi * 8;
  const _Float16* bp = Bp + (size_t)tn * 512 + lane * 16;
  size_t bstride = (size_t)Ntiles * 512;
  v8f c = {};
  v16h a = load_a16(ap);
  v16h b = *(const v16h*)bp;
  for (int k0 = 32; k0 < K; k0 += 32) {
    v16h an = load_a16(ap + k0);
    bp += bstride;
    v16h bn = *(const v16h*)bp;
    __builtin_prefetch(ap + k0 + 32, 0, 1);
    c = wmma_f16(a, b, c);
    a = an;
    b = bn;
  }
  c = wmma_f16(a, b, c);
  int ncol = tn * 16 + row;  // C layout: N = lane&15
  float bv = (ncol < nval) ? bias[ncol] : 0.f;
#pragma unroll
  for (int r = 0; r < 8; ++r) {
    int mm = tm * 16 + r + hi * 8;  // C layout: M = r + 8*hi
    float v = c[r] + bv;
    if (relu) v = fmaxf(v, 0.f);
    if (ncol < nval) {
      if (outF) outF[(size_t)mm * ldoF + ncol] = v;
      if (outH) outH[(size_t)mm * ldoH + ncol] = (_Float16)v;
    }
  }
}

// ---------------------------------------------------------------------------
// GRU gates (torch GRUCell semantics)
// ---------------------------------------------------------------------------
__global__ void gru_gates(const float* __restrict__ gi,
                          const float* __restrict__ gh,
                          const float* __restrict__ hprev,
                          float* __restrict__ newhF,
                          _Float16* __restrict__ newhH,
                          _Float16* __restrict__ combined) {
  int idx = blockIdx.x * blockDim.x + threadIdx.x;
  if (idx >= N_AG * HDIM) return;
  int h = idx & 127, n = idx >> 7;
  const float* gir = gi + (size_t)n * 384;
  const float* ghr = gh + (size_t)n * 384;
  float r  = 1.f / (1.f + __expf(-(gir[h] + ghr[h])));
  float z  = 1.f / (1.f + __expf(-(gir[128 + h] + ghr[128 + h])));
  float nn = tanhf(gir[256 + h] + r * ghr[256 + h]);
  float nh = (1.f - z) * nn + z * hprev[idx];
  newhF[idx] = nh;
  newhH[idx] = (_Float16)nh;
  combined[(size_t)n * 256 + h] = (_Float16)nh;
}

// ---------------------------------------------------------------------------
// Flash attention, cooperative version:
//  - 4 waves/block, one 16-query tile each, shared key stream.
//  - K/V fragments for each 32-key step are contiguous 8KB blocks of the
//    packed buffers; staged into double-buffered LDS via async-to-LDS copies
//    (ASYNCcnt) issued one step ahead, shared by all 4 waves.
//  - bit-packed adjacency mask, online softmax, probs staged via per-wave
//    LDS tile to reach the WMMA A-layout.
// ---------------------------------------------------------------------------
__global__ void flash_attn(const _Float16* __restrict__ Qh,
                           const _Float16* __restrict__ Kp,
                           const _Float16* __restrict__ Vp,
                           const unsigned int* __restrict__ amask,
                           _Float16* __restrict__ ctxH) {
  __shared__ __align__(16) _Float16 Kbuf[2][4096];  // 2 x 8KB
  __shared__ __align__(16) _Float16 Vbuf[2][4096];  // 2 x 8KB
  __shared__ __align__(16) _Float16 Pbuf[4][512];   // per-wave P tile
  int tid = threadIdx.x;
  int wave = tid >> 5, lane = tid & 31;
  int row = lane & 15, hi = lane >> 4;
  int qbase = (blockIdx.x * 4 + wave) * 16;
  int m = qbase + row;
  v16h qf[4];
#pragma unroll
  for (int kc = 0; kc < 4; ++kc)
    qf[kc] = load_a16(Qh + (size_t)m * HDIM + kc * 32 + hi * 8);
  v8f zero = {};
  v8f acc[8];
#pragma unroll
  for (int t = 0; t < 8; ++t) acc[t] = zero;
  float rmax[8], rsum[8];
#pragma unroll
  for (int r = 0; r < 8; ++r) { rmax[r] = -3.0e38f; rsum[r] = 0.f; }
  _Float16* pb = &Pbuf[wave][0];
  int ncl = lane & 15;

  // prologue: stage step 0 into buffer 0 (each thread copies 4x16B of K and V)
#pragma unroll
  for (int q = 0; q < 4; ++q) {
    copy_b128_to_lds(&Kbuf[0][(tid + 128 * q) * 8], Kp + (tid + 128 * q) * 8);
    copy_b128_to_lds(&Vbuf[0][(tid + 128 * q) * 8], Vp + (tid + 128 * q) * 8);
  }

  for (int j0 = 0; j0 < N_AG; j0 += 32) {
    int cur = (j0 >> 5) & 1;
    kv_copy_wait();    // my async copies into 'cur' complete
    __syncthreads();   // everyone staged; everyone done reading 'cur^1'
    if (j0 + 32 < N_AG) {
      const _Float16* gK = Kp + (size_t)(j0 + 32) * 128;
      const _Float16* gV = Vp + (size_t)(j0 + 32) * 128;
#pragma unroll
      for (int q = 0; q < 4; ++q) {
        copy_b128_to_lds(&Kbuf[cur ^ 1][(tid + 128 * q) * 8],
                         gK + (tid + 128 * q) * 8);
        copy_b128_to_lds(&Vbuf[cur ^ 1][(tid + 128 * q) * 8],
                         gV + (tid + 128 * q) * 8);
      }
    }
    const _Float16* lK = &Kbuf[cur][0];
    const _Float16* lV = &Vbuf[cur][0];
    // ---- scores for 32 keys: two 16x16 tiles, K-dim 128 in 4 WMMAs each
    v8f s0 = zero, s1 = zero;
#pragma unroll
    for (int kc = 0; kc < 4; ++kc) {
      v16h b0 = *(const v16h*)(lK + (size_t)kc * 512 + lane * 16);
      v16h b1 = *(const v16h*)(lK + (size_t)(4 + kc) * 512 + lane * 16);
      s0 = wmma_f16(qf[kc], b0, s0);
      s1 = wmma_f16(qf[kc], b1, s1);
    }
    // ---- V fragments issued early; LDS latency hides under softmax VALU
    v16h vs[8];
#pragma unroll
    for (int t = 0; t < 8; ++t)
      vs[t] = *(const v16h*)(lV + (size_t)t * 512 + lane * 16);
    // ---- mask + online softmax (row stats uniform per 16-lane half)
    int jv = j0 >> 5;
#pragma unroll
    for (int r = 0; r < 8; ++r) {
      int mm = qbase + r + hi * 8;
      unsigned int wbits = amask[(size_t)mm * 256 + jv];
      float v0 = s0[r] * SCALE_QK;
      v0 = ((wbits >> ncl) & 1u) ? v0 : -1e9f;
      float v1 = s1[r] * SCALE_QK;
      v1 = ((wbits >> (16 + ncl)) & 1u) ? v1 : -1e9f;
      float tmax = fmaxf(v0, v1);
#pragma unroll
      for (int off = 1; off < 16; off <<= 1)
        tmax = fmaxf(tmax, __shfl_xor(tmax, off, 32));
      float nm = fmaxf(rmax[r], tmax);
      float corr = __expf(rmax[r] - nm);
      rmax[r] = nm;
      float p0 = __expf(v0 - nm), p1 = __expf(v1 - nm);
      float ps = p0 + p1;
#pragma unroll
      for (int off = 1; off < 16; off <<= 1) ps += __shfl_xor(ps, off, 32);
      rsum[r] = rsum[r] * corr + ps;
#pragma unroll
      for (int t = 0; t < 8; ++t) acc[t][r] *= corr;
      int ml = r + hi * 8;
      pb[ml * 32 + ncl]      = (_Float16)p0;
      pb[ml * 32 + 16 + ncl] = (_Float16)p1;
    }
    // ---- context += P[16x32] * V[32x128]
    v16h aP = load_a16(pb + row * 32 + hi * 8);
#pragma unroll
    for (int t = 0; t < 8; ++t) acc[t] = wmma_f16(aP, vs[t], acc[t]);
  }
#pragma unroll
  for (int r = 0; r < 8; ++r) {
    float inv = 1.0f / rsum[r];
    int mm = qbase + r + hi * 8;
#pragma unroll
    for (int t = 0; t < 8; ++t)
      ctxH[(size_t)mm * HDIM + t * 16 + ncl] = (_Float16)(acc[t][r] * inv);
  }
}

// ---------------------------------------------------------------------------
// Host orchestration
// ---------------------------------------------------------------------------
extern "C" void kernel_launch(void* const* d_in, const int* in_sizes, int n_in,
                              void* d_out, int out_size, void* d_ws,
                              size_t ws_size, hipStream_t stream) {
  (void)in_sizes; (void)n_in; (void)out_size; (void)ws_size;
  const float* img  = (const float*)d_in[0];
  const int*   adj  = (const int*)d_in[1];
  const int*   cls  = (const int*)d_in[2];
  const float* hid  = (const float*)d_in[3];
  const float* w1   = (const float*)d_in[4];
  const float* b1   = (const float*)d_in[5];
  const float* w2   = (const float*)d_in[6];
  const float* b2   = (const float*)d_in[7];
  const float* fcvw = (const float*)d_in[8];
  const float* fcvb = (const float*)d_in[9];
  const float* emb  = (const float*)d_in[10];
  const float* wih  = (const float*)d_in[11];
  const float* whh  = (const float*)d_in[12];
  const float* bih  = (const float*)d_in[13];
  const float* bhh  = (const float*)d_in[14];
  const float* wq   = (const float*)d_in[15];
  const float* bq   = (const float*)d_in[16];
  const float* wk   = (const float*)d_in[17];
  const float* bk   = (const float*)d_in[18];
  const float* wv   = (const float*)d_in[19];
  const float* bv   = (const float*)d_in[20];
  const float* ao   = (const float*)d_in[21];
  const float* aob  = (const float*)d_in[22];
  const float* fcqw = (const float*)d_in[23];
  const float* fcqb = (const float*)d_in[24];

  float* out_q = (float*)d_out;             // [8192,5]
  float* out_h = (float*)d_out + N_AG * 5;  // [8192,128]

  char* ws = (char*)d_ws;
  auto alloc = [&](size_t bytes) -> char* {
    char* p = ws;
    ws += (bytes + 255) & ~(size_t)255;
    return p;
  };
  float*        c1o   = (float*)alloc((size_t)N_AG * 16 * 324 * 4);
  _Float16*     x2h   = (_Float16*)alloc((size_t)N_AG * 2048 * 2);
  _Float16*     feat  = (_Float16*)alloc((size_t)N_AG * 160 * 2);
  _Float16*     fcvP  = (_Float16*)alloc((size_t)64 * 8 * 512 * 2);
  _Float16*     wihP  = (_Float16*)alloc((size_t)5 * 24 * 512 * 2);
  _Float16*     whhP  = (_Float16*)alloc((size_t)4 * 24 * 512 * 2);
  _Float16*     wqP   = (_Float16*)alloc((size_t)4 * 8 * 512 * 2);
  _Float16*     wkP   = (_Float16*)alloc((size_t)4 * 8 * 512 * 2);
  _Float16*     wvP   = (_Float16*)alloc((size_t)4 * 8 * 512 * 2);
  _Float16*     aoP   = (_Float16*)alloc((size_t)4 * 8 * 512 * 2);
  _Float16*     fcqP  = (_Float16*)alloc((size_t)8 * 1 * 512 * 2);
  _Float16*     hidh  = (_Float16*)alloc((size_t)N_AG * 128 * 2);
  float*        gi    = (float*)alloc((size_t)N_AG * 384 * 4);
  float*        gh    = (float*)alloc((size_t)N_AG * 384 * 4);
  _Float16*     nhh   = (_Float16*)alloc((size_t)N_AG * 128 * 2);
  _Float16*     qh    = (_Float16*)alloc((size_t)N_AG * 128 * 2);
  _Float16*     kh    = (_Float16*)alloc((size_t)N_AG * 128 * 2);
  _Float16*     vh    = (_Float16*)alloc((size_t)N_AG * 128 * 2);
  _Float16*     kP    = (_Float16*)alloc((size_t)512 * 4 * 512 * 2);
  _Float16*     vP    = (_Float16*)alloc((size_t)256 * 8 * 512 * 2);
  unsigned int* amask = (unsigned int*)alloc((size_t)N_AG * 256 * 4);
  _Float16*     ctxh  = (_Float16*)alloc((size_t)N_AG * 128 * 2);
  _Float16*     comb  = (_Float16*)alloc((size_t)N_AG * 256 * 2);

  // CNN encoder
  conv1_relu<<<(N_AG * 16 * 324 + 255) / 256, 256, 0, stream>>>(img, w1, b1, c1o);
  conv2_relu_h<<<(N_AG * 2048 + 255) / 256, 256, 0, stream>>>(c1o, w2, b2, x2h);

  // Weight packing (convert + pad + B-fragment permute, one shot each)
  pack_wb<<<(64 * 8 * 512 + 255) / 256, 256, 0, stream>>>(fcvw, fcvP, 128, 2048, 64, 8);
  pack_wb<<<(5 * 24 * 512 + 255) / 256, 256, 0, stream>>>(wih, wihP, 384, 144, 5, 24);
  pack_wb<<<(4 * 24 * 512 + 255) / 256, 256, 0, stream>>>(whh, whhP, 384, 128, 4, 24);
  pack_wb<<<(4 * 8 * 512 + 255) / 256, 256, 0, stream>>>(wq, wqP, 128, 128, 4, 8);
  pack_wb<<<(4 * 8 * 512 + 255) / 256, 256, 0, stream>>>(wk, wkP, 128, 128, 4, 8);
  pack_wb<<<(4 * 8 * 512 + 255) / 256, 256, 0, stream>>>(wv, wvP, 128, 128, 4, 8);
  pack_wb<<<(4 * 8 * 512 + 255) / 256, 256, 0, stream>>>(ao, aoP, 128, 128, 4, 8);
  pack_wb<<<(8 * 1 * 512 + 255) / 256, 256, 0, stream>>>(fcqw, fcqP, 5, 256, 8, 1);
  f32_to_f16<<<(N_AG * 128 + 255) / 256, 256, 0, stream>>>(hid, hidh, N_AG * 128);
  pack_adj<<<(N_AG * 256 + 255) / 256, 256, 0, stream>>>(adj, amask);

  // fc_vis: [8192,2048] x [128,2048]^T, relu -> feat cols 0..127 (f16)
  gemm_xwt<<<(512 * 8) / 4, 128, 0, stream>>>(x2h, 2048, fcvP, fcvb,
                                              nullptr, 0, feat, 160,
                                              8, 2048, 128, 1);
  embed_fill<<<(N_AG * 32 + 255) / 256, 256, 0, stream>>>(emb, cls, feat);

  // GRU input/hidden GEMMs
  gemm_xwt<<<(512 * 24) / 4, 128, 0, stream>>>(feat, 160, wihP, bih,
                                               gi, 384, nullptr, 0,
                                               24, 160, 384, 0);
  gemm_xwt<<<(512 * 24) / 4, 128, 0, stream>>>(hidh, 128, whhP, bhh,
                                               gh, 384, nullptr, 0,
                                               24, 128, 384, 0);
  gru_gates<<<(N_AG * 128 + 255) / 256, 256, 0, stream>>>(gi, gh, hid, out_h,
                                                          nhh, comb);

  // Q, K, V projections (f16 outputs), then fragment packing for attention
  gemm_xwt<<<(512 * 8) / 4, 128, 0, stream>>>(nhh, 128, wqP, bq,
                                              nullptr, 0, qh, 128, 8, 128, 128, 0);
  gemm_xwt<<<(512 * 8) / 4, 128, 0, stream>>>(nhh, 128, wkP, bk,
                                              nullptr, 0, kh, 128, 8, 128, 128, 0);
  gemm_xwt<<<(512 * 8) / 4, 128, 0, stream>>>(nhh, 128, wvP, bv,
                                              nullptr, 0, vh, 128, 8, 128, 128, 0);
  pack_kb<<<(512 * 4 * 512 + 255) / 256, 256, 0, stream>>>(kh, kP);
  pack_vb<<<(256 * 8 * 512 + 255) / 256, 256, 0, stream>>>(vh, vP);

  // Masked flash attention: 512 query tiles, 4 waves/block
  flash_attn<<<N_AG / 64, 128, 0, stream>>>(qh, kP, vP, amask, ctxh);

  // attn_out projection -> combined cols 128..255
  gemm_xwt<<<(512 * 8) / 4, 128, 0, stream>>>(ctxh, 128, aoP, aob,
                                              nullptr, 0, comb + 128, 256,
                                              8, 128, 128, 0);

  // Q-head: [8192,256] x [5,256]^T -> d_out (N padded to one 16-tile)
  gemm_xwt<<<512 / 4, 128, 0, stream>>>(comb, 256, fcqP, fcqb,
                                        out_q, 5, nullptr, 0, 1, 256, 5, 0);
}
